// HMAEVIT_72730976190642
// MI455X (gfx1250) — matrix operand order repaired and checked by reading, not compiled, not run on hardware
//
#include <hip/hip_runtime.h>
#include <hip/hip_bf16.h>
#include <math.h>

typedef __attribute__((ext_vector_type(16))) _Float16 v16h;
typedef __attribute__((ext_vector_type(8)))  _Float16 v8h;
typedef __attribute__((ext_vector_type(8)))  float    v8f;

#define BM 128
#define BN 128
#define BK 32
#define LDA 40   // halves per A-tile row (padded)
#define LDB 40   // halves per B^T-tile row (padded)

__device__ __forceinline__ float gelu_exact(float x) {
    return 0.5f * x * (1.0f + erff(x * 0.70710678118654752f));
}

// C[M,N] = act(A[M,K] @ W[K,N] + bias[N]) (+ resid[M,N] if non-null)
// M % 128 == 0, N % 128 == 0, K % 32 == 0 (guaranteed by the model dims).
// 256 threads = 8 waves; each wave owns a 16x128 strip (8 WMMA accumulators).
// Software-pipelined: next K-tile is loaded into registers while WMMAs run.
__global__ __launch_bounds__(256) void hmae_gemm_wmma(
    const float* __restrict__ A, const float* __restrict__ W,
    const float* __restrict__ bias, const float* __restrict__ resid,
    float* __restrict__ C, int M, int N, int K, int act)
{
    __shared__ _Float16 As[BM * LDA];
    __shared__ _Float16 Bs[BN * LDB];

    const int t     = threadIdx.x;
    const int wave  = t >> 5;
    const int lane  = t & 31;
    const int ln16  = lane & 15;
    const int lhalf = (lane >> 4) & 1;
    const int m0    = blockIdx.y * BM;
    const int nblk  = blockIdx.x * BN;
    const int mw    = wave * 16;

    v8f acc[8] = {v8f{}, v8f{}, v8f{}, v8f{}, v8f{}, v8f{}, v8f{}, v8f{}};

    // A staging: 2 threads per row, 16 consecutive floats each
    const int arow = t >> 1;
    const int aseg = (t & 1) * 16;
    // B staging (transposed into Bs[n][k]): 8 threads per k-row, 16 cols each
    const int bk  = t >> 3;
    const int bn0 = (t & 7) * 16;

    const float* agbase = A + (size_t)(m0 + arow) * K + aseg;
    const float* wgbase = W + (size_t)bk * N + nblk + bn0;
    const size_t wstep  = (size_t)BK * N;

    float areg[16];
    float breg[16];

    // Prologue: stage K-tile 0
    #pragma unroll
    for (int i = 0; i < 16; ++i) areg[i] = agbase[i];
    #pragma unroll
    for (int i = 0; i < 16; ++i) breg[i] = wgbase[i];
    {
        _Float16* as = &As[arow * LDA + aseg];
        #pragma unroll
        for (int i = 0; i < 16; ++i) as[i] = (_Float16)areg[i];
        #pragma unroll
        for (int i = 0; i < 16; ++i) Bs[(bn0 + i) * LDB + bk] = (_Float16)breg[i];
    }
    __syncthreads();

    for (int k0 = 0; k0 < K; k0 += BK) {
        const bool more = (k0 + BK) < K;
        if (more) {
            // Issue next tile's global loads; they retire while WMMAs execute.
            const float* ag = agbase + k0 + BK;
            const float* wg = wgbase + (size_t)(k0 + BK) * N;
            #pragma unroll
            for (int i = 0; i < 16; ++i) areg[i] = ag[i];
            #pragma unroll
            for (int i = 0; i < 16; ++i) breg[i] = wg[i];
            if (k0 + 2 * BK < K) {
                __builtin_prefetch(agbase + k0 + 2 * BK, 0, 3);
                __builtin_prefetch(wgbase + (size_t)(k0 + 2 * BK) * N, 0, 3);
            }
        }

        // A fragment: 16-bit A 16x32 layout (lane<16: K 0-7 / 16-23; lane>=16: +8)
        const _Float16* ap = &As[(mw + ln16) * LDA + lhalf * 8];
        v8h alo = *(const v8h*)ap;
        v8h ahi = *(const v8h*)(ap + 16);
        v16h afrag = __builtin_shufflevector(alo, ahi,
            0,1,2,3,4,5,6,7,8,9,10,11,12,13,14,15);

        #pragma unroll
        for (int nt = 0; nt < 8; ++nt) {
            // B fragment: lane = N, K 0-7 / 16-23 (lane>=16: +8), matching A
            const _Float16* bp = &Bs[(nt * 16 + ln16) * LDB + lhalf * 8];
            v8h blo = *(const v8h*)bp;
            v8h bhi = *(const v8h*)(bp + 16);
            v16h bfrag = __builtin_shufflevector(blo, bhi,
                0,1,2,3,4,5,6,7,8,9,10,11,12,13,14,15);
            acc[nt] = __builtin_amdgcn_wmma_f32_16x16x32_f16(
                false, afrag, false, bfrag, (short)0, acc[nt], false, false);
        }
        __syncthreads();

        if (more) {
            _Float16* as = &As[arow * LDA + aseg];
            #pragma unroll
            for (int i = 0; i < 16; ++i) as[i] = (_Float16)areg[i];
            #pragma unroll
            for (int i = 0; i < 16; ++i) Bs[(bn0 + i) * LDB + bk] = (_Float16)breg[i];
        }
        __syncthreads();
    }

    // Epilogue. C/D layout: VGPR r -> M=r (lanes 0-15), M=r+8 (lanes 16-31); N = lane&15
    #pragma unroll
    for (int nt = 0; nt < 8; ++nt) {
        const int col = nblk + nt * 16 + ln16;
        const float bv = bias[col];
        #pragma unroll
        for (int r = 0; r < 8; ++r) {
            const int row = m0 + mw + lhalf * 8 + r;
            const size_t idx = (size_t)row * N + col;
            float v = acc[nt][r] + bv;
            if (act) v = gelu_exact(v);
            if (resid) v += resid[idx];
            C[idx] = v;
        }
    }
}

// LayerNorm per token; optional mask -> write zeros for masked tokens.
__global__ __launch_bounds__(256) void hmae_ln(
    const float* __restrict__ x, const float* __restrict__ w, const float* __restrict__ b,
    float* __restrict__ y, int D, const int* __restrict__ mask)
{
    __shared__ float red[256];
    const int tk = blockIdx.x;
    const float* xr = x + (size_t)tk * D;
    float s = 0.f;
    for (int d = threadIdx.x; d < D; d += 256) s += xr[d];
    red[threadIdx.x] = s; __syncthreads();
    for (int o = 128; o > 0; o >>= 1) {
        if (threadIdx.x < o) red[threadIdx.x] += red[threadIdx.x + o];
        __syncthreads();
    }
    const float mean = red[0] / (float)D;
    __syncthreads();
    float v = 0.f;
    for (int d = threadIdx.x; d < D; d += 256) { float e = xr[d] - mean; v += e * e; }
    red[threadIdx.x] = v; __syncthreads();
    for (int o = 128; o > 0; o >>= 1) {
        if (threadIdx.x < o) red[threadIdx.x] += red[threadIdx.x + o];
        __syncthreads();
    }
    const float rstd = rsqrtf(red[0] / (float)D + 1e-6f);
    const int zero = (mask != nullptr) && (mask[tk] != 0);
    float* yr = y + (size_t)tk * D;
    for (int d = threadIdx.x; d < D; d += 256)
        yr[d] = zero ? 0.f : (xr[d] - mean) * rstd * w[d] + b[d];
}

// Flash-style attention: one wave (32 lanes) per query row, online softmax.
// qkv[t][3D]: q = [h*hd..], k = [D + h*hd..], v = [2D + h*hd..]. hd in {32, 64}.
__global__ __launch_bounds__(256) void hmae_attn(
    const float* __restrict__ qkv, float* __restrict__ out,
    const int* __restrict__ mask, int B, int L, int nh, int hd, int D)
{
    const int wave = threadIdx.x >> 5;
    const int lane = threadIdx.x & 31;
    const int gq = blockIdx.x * 8 + wave;
    const int q  = gq % L;
    const int bh = gq / L;
    const int h  = bh % nh;
    const int b  = bh / nh;
    const int nd = hd >> 5;           // 1 (hd=32) or 2 (hd=64)
    const float scale = rsqrtf((float)hd);
    const size_t row3 = (size_t)3 * D;

    const float* qp = qkv + (size_t)(b * L + q) * row3 + h * hd;
    float q0 = qp[lane] * scale;
    float q1 = (nd == 2) ? qp[lane + 32] * scale : 0.f;

    const float* kb = qkv + (size_t)(b * L) * row3 + D + h * hd;
    const float* vb = kb + D;
    const int* mrow = (mask != nullptr) ? (mask + b * L) : nullptr;

    float m = -3.0e38f, l = 0.f, o0 = 0.f, o1 = 0.f;
    for (int kt = 0; kt < L; ++kt) {
        const float* kp = kb + (size_t)kt * row3;
        float p = q0 * kp[lane];
        if (nd == 2) p += q1 * kp[lane + 32];
        #pragma unroll
        for (int off = 16; off > 0; off >>= 1) p += __shfl_xor(p, off, 32);
        float s = p;
        if (mrow && mrow[kt] != 0) s = -1e9f;
        const float mn = fmaxf(m, s);
        const float co = expf(m - mn);
        const float pe = expf(s - mn);
        const float* vp = vb + (size_t)kt * row3;
        o0 = o0 * co + pe * vp[lane];
        if (nd == 2) o1 = o1 * co + pe * vp[lane + 32];
        l = l * co + pe;
        m = mn;
    }
    float* op = out + (size_t)(b * L + q) * D + h * hd;
    const float inv = 1.f / l;
    op[lane] = o0 * inv;
    if (nd == 2) op[lane + 32] = o1 * inv;
}

// Positional MLP: x[t] += gelu(f(t) @ w1 + b1) @ w2 + b2, f = 5 box features.
__global__ __launch_bounds__(256) void hmae_pos(
    const float* __restrict__ coords, const float* __restrict__ depths,
    const float* __restrict__ w1, const float* __restrict__ b1,
    const float* __restrict__ w2, const float* __restrict__ b2,
    float* __restrict__ x, int H, int D)
{
    __shared__ float hid[384];
    const int tk = blockIdx.x;
    const float* c = coords + (size_t)tk * 4;
    const float x1 = c[0], x2 = c[1], y1 = c[2], y2 = c[3];
    const float f0 = (x1 + x2) * 0.5f * (1.f / 1024.f);
    const float f1 = (y1 + y2) * 0.5f * (1.f / 1024.f);
    const float f2 = (x2 - x1) * (1.f / 1024.f);
    const float f3 = (y2 - y1) * (1.f / 1024.f);
    const float f4 = depths[tk] * (1.f / 8.f);
    for (int j = threadIdx.x; j < H; j += 256) {
        float a = b1[j] + f0 * w1[0 * H + j] + f1 * w1[1 * H + j]
                + f2 * w1[2 * H + j] + f3 * w1[3 * H + j] + f4 * w1[4 * H + j];
        hid[j] = gelu_exact(a);
    }
    __syncthreads();
    float* xr = x + (size_t)tk * D;
    for (int d = threadIdx.x; d < D; d += 256) {
        float a = b2[d];
        for (int j = 0; j < H; ++j) a += hid[j] * w2[(size_t)j * D + d];
        xr[d] += a;
    }
}

__global__ __launch_bounds__(256) void hmae_mask_token(
    float* __restrict__ xd, const int* __restrict__ mask,
    const float* __restrict__ tok, int DD, int total)
{
    const int i = blockIdx.x * blockDim.x + threadIdx.x;
    if (i >= total) return;
    const int t = i / DD, d = i - t * DD;
    if (mask[t] != 0) xd[i] = tok[d];
}

extern "C" void kernel_launch(void* const* d_in, const int* in_sizes, int n_in,
                              void* d_out, int out_size, void* d_ws, size_t ws_size,
                              hipStream_t stream) {
    (void)in_sizes; (void)n_in; (void)out_size; (void)ws_size;
    const float* patches   = (const float*)d_in[0];
    const float* coords    = (const float*)d_in[1];
    const float* depths    = (const float*)d_in[2];
    const int*   mask      = (const int*)d_in[3];
    const float* pe_w      = (const float*)d_in[4];
    const float* pe_b      = (const float*)d_in[5];
    const float* pos1_w    = (const float*)d_in[6];
    const float* pos1_b    = (const float*)d_in[7];
    const float* pos2_w    = (const float*)d_in[8];
    const float* pos2_b    = (const float*)d_in[9];
    const float* e_ln1_w   = (const float*)d_in[10];
    const float* e_ln1_b   = (const float*)d_in[11];
    const float* e_qkv_w   = (const float*)d_in[12];
    const float* e_qkv_b   = (const float*)d_in[13];
    const float* e_pr_w    = (const float*)d_in[14];
    const float* e_pr_b    = (const float*)d_in[15];
    const float* e_ln2_w   = (const float*)d_in[16];
    const float* e_ln2_b   = (const float*)d_in[17];
    const float* e_f1_w    = (const float*)d_in[18];
    const float* e_f1_b    = (const float*)d_in[19];
    const float* e_f2_w    = (const float*)d_in[20];
    const float* e_f2_b    = (const float*)d_in[21];
    const float* enorm_w   = (const float*)d_in[22];
    const float* enorm_b   = (const float*)d_in[23];
    const float* de_w      = (const float*)d_in[24];
    const float* de_b      = (const float*)d_in[25];
    const float* mask_tok  = (const float*)d_in[26];
    const float* dpos1_w   = (const float*)d_in[27];
    const float* dpos1_b   = (const float*)d_in[28];
    const float* dpos2_w   = (const float*)d_in[29];
    const float* dpos2_b   = (const float*)d_in[30];
    const float* d_ln1_w   = (const float*)d_in[31];
    const float* d_ln1_b   = (const float*)d_in[32];
    const float* d_qkv_w   = (const float*)d_in[33];
    const float* d_qkv_b   = (const float*)d_in[34];
    const float* d_pr_w    = (const float*)d_in[35];
    const float* d_pr_b    = (const float*)d_in[36];
    const float* d_ln2_w   = (const float*)d_in[37];
    const float* d_ln2_b   = (const float*)d_in[38];
    const float* d_f1_w    = (const float*)d_in[39];
    const float* d_f1_b    = (const float*)d_in[40];
    const float* d_f2_w    = (const float*)d_in[41];
    const float* d_f2_b    = (const float*)d_in[42];
    const float* dnorm_w   = (const float*)d_in[43];
    const float* dnorm_b   = (const float*)d_in[44];
    const float* hi_w      = (const float*)d_in[45];
    const float* hi_b      = (const float*)d_in[46];
    const float* hn_w      = (const float*)d_in[47];
    const float* hn_b      = (const float*)d_in[48];

    constexpr int B = 4, L = 512, D = 768, DD = 512, ED = 12, DDEP = 8, PD = 1024;
    constexpr int M = B * L;  // 2048 tokens

    // Workspace layout (floats), 40 MB total:
    // xe [M*D], lnb [M*D] (LN out + attn out), big [M*4D] (qkv & MLP hidden), xd [M*DD]
    float* xe  = (float*)d_ws;
    float* lnb = xe  + (size_t)M * D;
    float* big = lnb + (size_t)M * D;
    float* xd  = big + (size_t)M * 4 * D;
    float* out = (float*)d_out;

    auto gemm = [&](const float* A, const float* W, const float* bias,
                    const float* resid, float* C, int Mm, int Nn, int Kk, int act) {
        dim3 g(Nn / BN, Mm / BM);
        hipLaunchKernelGGL(hmae_gemm_wmma, g, dim3(256), 0, stream,
                           A, W, bias, resid, C, Mm, Nn, Kk, act);
    };

    // ---- Patch embed + encoder positional MLP ----
    gemm(patches, pe_w, pe_b, nullptr, xe, M, D, PD, 0);
    hipLaunchKernelGGL(hmae_pos, dim3(M), dim3(256), 0, stream,
                       coords, depths, pos1_w, pos1_b, pos2_w, pos2_b, xe, D / 2, D);

    // ---- Encoder: 12 blocks, 12 heads, key-mask = visible ----
    for (int i = 0; i < ED; ++i) {
        hipLaunchKernelGGL(hmae_ln, dim3(M), dim3(256), 0, stream,
                           xe, e_ln1_w + (size_t)i * D, e_ln1_b + (size_t)i * D, lnb, D, (const int*)nullptr);
        gemm(lnb, e_qkv_w + (size_t)i * D * 3 * D, e_qkv_b + (size_t)i * 3 * D,
             nullptr, big, M, 3 * D, D, 0);
        hipLaunchKernelGGL(hmae_attn, dim3(B * 12 * L / 8), dim3(256), 0, stream,
                           big, lnb, mask, B, L, 12, 64, D);
        gemm(lnb, e_pr_w + (size_t)i * D * D, e_pr_b + (size_t)i * D, xe, xe, M, D, D, 0);
        hipLaunchKernelGGL(hmae_ln, dim3(M), dim3(256), 0, stream,
                           xe, e_ln2_w + (size_t)i * D, e_ln2_b + (size_t)i * D, lnb, D, (const int*)nullptr);
        gemm(lnb, e_f1_w + (size_t)i * D * 4 * D, e_f1_b + (size_t)i * 4 * D,
             nullptr, big, M, 4 * D, D, 1);
        gemm(big, e_f2_w + (size_t)i * 4 * D * D, e_f2_b + (size_t)i * D, xe, xe, M, D, 4 * D, 0);
    }

    // ---- Encoder norm (zero masked tokens), decoder embed, mask token, dec pos ----
    hipLaunchKernelGGL(hmae_ln, dim3(M), dim3(256), 0, stream,
                       xe, enorm_w, enorm_b, lnb, D, mask);
    gemm(lnb, de_w, de_b, nullptr, xd, M, DD, D, 0);
    hipLaunchKernelGGL(hmae_mask_token, dim3((M * DD + 255) / 256), dim3(256), 0, stream,
                       xd, mask, mask_tok, DD, M * DD);
    hipLaunchKernelGGL(hmae_pos, dim3(M), dim3(256), 0, stream,
                       coords, depths, dpos1_w, dpos1_b, dpos2_w, dpos2_b, xd, DD / 2, DD);

    // ---- Decoder: 8 blocks, 16 heads, no mask ----
    for (int i = 0; i < DDEP; ++i) {
        hipLaunchKernelGGL(hmae_ln, dim3(M), dim3(256), 0, stream,
                           xd, d_ln1_w + (size_t)i * DD, d_ln1_b + (size_t)i * DD, lnb, DD, (const int*)nullptr);
        gemm(lnb, d_qkv_w + (size_t)i * DD * 3 * DD, d_qkv_b + (size_t)i * 3 * DD,
             nullptr, big, M, 3 * DD, DD, 0);
        hipLaunchKernelGGL(hmae_attn, dim3(B * 16 * L / 8), dim3(256), 0, stream,
                           big, lnb, (const int*)nullptr, B, L, 16, 32, DD);
        gemm(lnb, d_pr_w + (size_t)i * DD * DD, d_pr_b + (size_t)i * DD, xd, xd, M, DD, DD, 0);
        hipLaunchKernelGGL(hmae_ln, dim3(M), dim3(256), 0, stream,
                           xd, d_ln2_w + (size_t)i * DD, d_ln2_b + (size_t)i * DD, lnb, DD, (const int*)nullptr);
        gemm(lnb, d_f1_w + (size_t)i * DD * 4 * DD, d_f1_b + (size_t)i * 4 * DD,
             nullptr, big, M, 4 * DD, DD, 1);
        gemm(big, d_f2_w + (size_t)i * 4 * DD * DD, d_f2_b + (size_t)i * DD, xd, xd, M, DD, 4 * DD, 0);
    }

    // ---- Final norm + two prediction heads (concatenated into d_out) ----
    hipLaunchKernelGGL(hmae_ln, dim3(M), dim3(256), 0, stream,
                       xd, dnorm_w, dnorm_b, lnb, DD, (const int*)nullptr);
    gemm(lnb, hi_w, hi_b, nullptr, out, M, PD, DD, 0);
    gemm(lnb, hn_w, hn_b, nullptr, out + (size_t)M * PD, M, PD, DD, 0);
}
